// MLPMoe_baseline_71390946394704
// MI455X (gfx1250) — matrix operand order repaired and checked by reading, not compiled, not run on hardware
//
#include <hip/hip_runtime.h>

typedef __attribute__((ext_vector_type(16))) __bf16 v16bf;
typedef __attribute__((ext_vector_type(8)))  __bf16 bf16x8;
typedef __attribute__((ext_vector_type(4)))  __bf16 bf16x4;
typedef __attribute__((ext_vector_type(8)))  float  v8f;

#define N_TOKENS  8192
#define D_MODEL   1024
#define N_EXPERTS 8

// GEMM tiling
#define BM 128
#define BN 64
#define BK 32
#define LDK 40   // padded LDS row stride in bf16 elems (80B = 20 dwords -> r*20 mod 64 distinct for r=0..15)

// workspace layout (bytes)
#define WS_CNT     0
#define WS_IDX     1024
#define WS_W       (1024 + N_EXPERTS * N_TOKENS * 4)
#define WS_PARTIAL (1024 + 2 * N_EXPERTS * N_TOKENS * 4)

// ---------------------------------------------------------------------------
__global__ __launch_bounds__(32)
void zero_cnt(int* __restrict__ cnt) {
    if (threadIdx.x < N_EXPERTS) cnt[threadIdx.x] = 0;
}

// ---------------------------------------------------------------------------
// Router: one wave32 per token. Dots vs 8 gate rows, wave reduction, top-2 on
// logits (== top-2 on softmax probs), renormalized weights, compaction.
__global__ __launch_bounds__(256)
void moe_router(const float* __restrict__ x, const float* __restrict__ gw,
                float* __restrict__ logits_out, int* __restrict__ cnt,
                int* __restrict__ rowIdx, float* __restrict__ rowW)
{
    const int lane = threadIdx.x & 31;
    const int wave = threadIdx.x >> 5;
    const int n    = blockIdx.x * 8 + wave;

    const float4* xr = (const float4*)(x + (size_t)n * D_MODEL);
    float acc[N_EXPERTS];
#pragma unroll
    for (int e = 0; e < N_EXPERTS; ++e) acc[e] = 0.f;

#pragma unroll
    for (int it = 0; it < D_MODEL / 4 / 32; ++it) {   // 8 iterations
        const int d4 = it * 32 + lane;
        const float4 xv = xr[d4];
#pragma unroll
        for (int e = 0; e < N_EXPERTS; ++e) {
            const float4 gv = ((const float4*)(gw + e * D_MODEL))[d4];
            acc[e] += xv.x * gv.x + xv.y * gv.y + xv.z * gv.z + xv.w * gv.w;
        }
    }
#pragma unroll
    for (int e = 0; e < N_EXPERTS; ++e) {
#pragma unroll
        for (int off = 16; off > 0; off >>= 1)
            acc[e] += __shfl_xor(acc[e], off, 32);
    }

    if (lane == 0) {
#pragma unroll
        for (int e = 0; e < N_EXPERTS; ++e)
            logits_out[(size_t)n * N_EXPERTS + e] = acc[e];

        // top-1 (strict > keeps lowest index on ties, matching jax top_k)
        int b0 = 0; float l0 = acc[0];
#pragma unroll
        for (int e = 1; e < N_EXPERTS; ++e)
            if (acc[e] > l0) { l0 = acc[e]; b0 = e; }
        // top-2
        int b1 = (b0 == 0) ? 1 : 0; float l1 = acc[b1];
#pragma unroll
        for (int e = 0; e < N_EXPERTS; ++e)
            if (e != b0 && acc[e] > l1) { l1 = acc[e]; b1 = e; }

        // renormalized softmax over the two selected logits
        const float w0 = 1.f / (1.f + __expf(l1 - l0));
        const float w1 = 1.f - w0;

        const int p0 = atomicAdd(&cnt[b0], 1);
        rowIdx[b0 * N_TOKENS + p0] = n * 2 + 0;
        rowW  [b0 * N_TOKENS + p0] = w0;
        const int p1 = atomicAdd(&cnt[b1], 1);
        rowIdx[b1 * N_TOKENS + p1] = n * 2 + 1;
        rowW  [b1 * N_TOKENS + p1] = w1;
    }
}

// ---------------------------------------------------------------------------
// Expert GEMM: per (expert, 128-row tile of its compacted list, 64-col tile).
// fp32 -> bf16 hi/lo split in LDS; 3x v_wmma_f32_16x16x32_bf16 per K-step per
// C-tile (a_hi*b_hi + a_hi*b_lo + a_lo*b_hi) ~ fp32 accuracy.
// Global loads for tile k+1 are software-pipelined through registers so they
// overlap the DS-load + WMMA compute phase of tile k.
__global__ __launch_bounds__(256)
void moe_expert_gemm(const float* __restrict__ x,
                     const float* __restrict__ w,     // [E][H][D]
                     const float* __restrict__ bias,  // [E][H]
                     const int*   __restrict__ cnt,
                     const int*   __restrict__ rowIdx,
                     const float* __restrict__ rowW,
                     float* __restrict__ partial)     // [2][N][H]
{
    const int e       = blockIdx.z;
    const int rowBase = blockIdx.y * BM;
    const int colBase = blockIdx.x * BN;
    const int count   = cnt[e];
    if (rowBase >= count) return;

    __shared__ alignas(16) __bf16 sAhi[BM][LDK];
    __shared__ alignas(16) __bf16 sAlo[BM][LDK];
    __shared__ alignas(16) __bf16 sBhi[BN][LDK];
    __shared__ alignas(16) __bf16 sBlo[BN][LDK];
    __shared__ int   sIdx[BM];
    __shared__ float sW[BM];

    const int tid = threadIdx.x;
    if (tid < BM) {
        const int r = rowBase + tid;
        if (r < count) { sIdx[tid] = rowIdx[e * N_TOKENS + r]; sW[tid] = rowW[e * N_TOKENS + r]; }
        else           { sIdx[tid] = -1;                       sW[tid] = 0.f; }
    }
    __syncthreads();

    // staging assignments
    const int ar = tid >> 1;            // A row 0..127 (two threads per row)
    const int ak = (tid & 1) * 16;      // 16 consecutive K floats
    const int code0 = sIdx[ar];
    const float* aSrc = x + (size_t)((code0 >= 0) ? (code0 >> 1) : 0) * D_MODEL + ak;
    const int br = tid >> 2;            // B row (output column) 0..63
    const int bk = (tid & 3) * 8;       // 8 consecutive K floats
    const float* bSrc = w + ((size_t)e * D_MODEL + (size_t)(colBase + br)) * D_MODEL + bk;

    const int lane = tid & 31;
    const int wave = tid >> 5;

    // register staging buffers for the software pipeline
    float4 aReg[4];
    float4 bReg[2];

    auto loadTile = [&](int k0) {
#pragma unroll
        for (int j = 0; j < 4; ++j) aReg[j] = *(const float4*)(aSrc + k0 + j * 4);
#pragma unroll
        for (int j = 0; j < 2; ++j) bReg[j] = *(const float4*)(bSrc + k0 + j * 4);
    };

    auto storeTile = [&]() {
#pragma unroll
        for (int j = 0; j < 4; ++j) {
            const float f[4] = {aReg[j].x, aReg[j].y, aReg[j].z, aReg[j].w};
            bf16x4 hv, lv;
#pragma unroll
            for (int q = 0; q < 4; ++q) {
                const __bf16 h = (__bf16)f[q];
                hv[q] = h;
                lv[q] = (__bf16)(f[q] - (float)h);
            }
            *(bf16x4*)&sAhi[ar][ak + j * 4] = hv;
            *(bf16x4*)&sAlo[ar][ak + j * 4] = lv;
        }
#pragma unroll
        for (int j = 0; j < 2; ++j) {
            const float f[4] = {bReg[j].x, bReg[j].y, bReg[j].z, bReg[j].w};
            bf16x4 hv, lv;
#pragma unroll
            for (int q = 0; q < 4; ++q) {
                const __bf16 h = (__bf16)f[q];
                hv[q] = h;
                lv[q] = (__bf16)(f[q] - (float)h);
            }
            *(bf16x4*)&sBhi[br][bk + j * 4] = hv;
            *(bf16x4*)&sBlo[br][bk + j * 4] = lv;
        }
    };

    v8f acc[4] = {};

    // fragment addressing (constant across K-steps)
    const int am  = (wave << 4) + (lane & 15);
    const int akb = (lane >> 4) << 4;   // A: 0 or 16  (K 0..15 / 16..31)
    const int bn  = lane & 15;
    const int bkb = (lane >> 4) << 4;   // B: 0 or 16  (K 0..15 / 16..31)

    loadTile(0);                         // prologue

    for (int k0 = 0; k0 < D_MODEL; k0 += BK) {
        storeTile();                     // convert + LDS-store tile k0 (from regs)
        __syncthreads();

        if (k0 + BK < D_MODEL)
            loadTile(k0 + BK);           // prefetch tile k0+BK; overlaps compute below

        // ---- A fragments (16x32 bf16)
        const bf16x8 ah0 = *(const bf16x8*)&sAhi[am][akb];
        const bf16x8 ah1 = *(const bf16x8*)&sAhi[am][akb + 8];
        const bf16x8 al0 = *(const bf16x8*)&sAlo[am][akb];
        const bf16x8 al1 = *(const bf16x8*)&sAlo[am][akb + 8];
        const v16bf ahi = __builtin_shufflevector(ah0, ah1, 0,1,2,3,4,5,6,7,8,9,10,11,12,13,14,15);
        const v16bf alo = __builtin_shufflevector(al0, al1, 0,1,2,3,4,5,6,7,8,9,10,11,12,13,14,15);

        // ---- B fragments (32x16 bf16) + WMMA
#pragma unroll
        for (int t = 0; t < 4; ++t) {
            const int rb = t * 16 + bn;
            const bf16x8 bh0 = *(const bf16x8*)&sBhi[rb][bkb];
            const bf16x8 bh1 = *(const bf16x8*)&sBhi[rb][bkb + 8];
            const bf16x8 bl0 = *(const bf16x8*)&sBlo[rb][bkb];
            const bf16x8 bl1 = *(const bf16x8*)&sBlo[rb][bkb + 8];
            const v16bf bhi = __builtin_shufflevector(bh0, bh1, 0,1,2,3,4,5,6,7,8,9,10,11,12,13,14,15);
            const v16bf blo = __builtin_shufflevector(bl0, bl1, 0,1,2,3,4,5,6,7,8,9,10,11,12,13,14,15);

            acc[t] = __builtin_amdgcn_wmma_f32_16x16x32_bf16(false, ahi, false, bhi, (short)0, acc[t], false, false);
            acc[t] = __builtin_amdgcn_wmma_f32_16x16x32_bf16(false, ahi, false, blo, (short)0, acc[t], false, false);
            acc[t] = __builtin_amdgcn_wmma_f32_16x16x32_bf16(false, alo, false, bhi, (short)0, acc[t], false, false);
        }
        __syncthreads();
    }

    // ---- epilogue: D layout: VGPR i -> M = i + 8*(lane/16), N = lane%16
    const int mOff = (lane >> 4) << 3;
    const int nOff = lane & 15;
#pragma unroll
    for (int t = 0; t < 4; ++t) {
        const int col = colBase + t * 16 + nOff;
        const float bv = bias[(size_t)e * D_MODEL + col];
#pragma unroll
        for (int i = 0; i < 8; ++i) {
            const int rloc = (wave << 4) + mOff + i;
            const int code = sIdx[rloc];
            if (code >= 0) {
                const int n    = code >> 1;
                const int slot = code & 1;
                partial[((size_t)slot * N_TOKENS + (size_t)n) * D_MODEL + col] =
                    sW[rloc] * (acc[t][i] + bv);
            }
        }
    }
}

// ---------------------------------------------------------------------------
__global__ __launch_bounds__(256)
void moe_combine(const float* __restrict__ p, float* __restrict__ out)
{
    const size_t i = (size_t)blockIdx.x * 256 + threadIdx.x;
    const float4 a = ((const float4*)p)[i];
    const float4 b = ((const float4*)(p + (size_t)N_TOKENS * D_MODEL))[i];
    float4 r;
    r.x = a.x + b.x; r.y = a.y + b.y; r.z = a.z + b.z; r.w = a.w + b.w;
    ((float4*)out)[i] = r;
}

// ---------------------------------------------------------------------------
extern "C" void kernel_launch(void* const* d_in, const int* in_sizes, int n_in,
                              void* d_out, int out_size, void* d_ws, size_t ws_size,
                              hipStream_t stream)
{
    (void)in_sizes; (void)n_in; (void)out_size; (void)ws_size;

    const float* x  = (const float*)d_in[0];
    const float* gw = (const float*)d_in[1];
    const float* ew = (const float*)d_in[2];
    const float* eb = (const float*)d_in[3];

    float* out    = (float*)d_out;
    float* logits = out + (size_t)N_TOKENS * D_MODEL;  // second tuple output

    char*  ws      = (char*)d_ws;
    int*   cnt     = (int*)(ws + WS_CNT);
    int*   rowIdx  = (int*)(ws + WS_IDX);
    float* rowW    = (float*)(ws + WS_W);
    float* partial = (float*)(ws + WS_PARTIAL);

    zero_cnt<<<1, 32, 0, stream>>>(cnt);
    moe_router<<<N_TOKENS / 8, 256, 0, stream>>>(x, gw, logits, cnt, rowIdx, rowW);
    moe_expert_gemm<<<dim3(D_MODEL / BN, N_TOKENS / BM, N_EXPERTS), 256, 0, stream>>>(
        x, ew, eb, cnt, rowIdx, rowW, partial);
    moe_combine<<<(N_TOKENS * D_MODEL / 4) / 256, 256, 0, stream>>>(partial, out);
}
